// CostGlobalEncoder_21766894256788
// MI455X (gfx1250) — matrix operand order repaired and checked by reading, not compiled, not run on hardware
//
#include <hip/hip_runtime.h>
#include <hip/hip_bf16.h>

// ---------------- problem constants ----------------
#define NB    8
#define HH    48
#define WWID  64
#define HWP   3072         // H*W
#define DCH   128
#define CIN   324
#define CPAD  352          // 324 padded to multiple of 32
#define KTAPS 9
#define HPAD  50           // 48 + halo
#define WPAD  66           // 64 + halo
#define KROW  40           // padded LDS row stride for 32-col tiles (bank spread)
#define DROW  136          // padded LDS row stride for 128-col tiles

typedef __attribute__((ext_vector_type(16))) __bf16 bf16x16;
typedef __attribute__((ext_vector_type(8)))  float  f32x8;
typedef unsigned short u16t;

union Frag { bf16x16 v; u16t u[16]; uint4 q[2]; };

__device__ __forceinline__ u16t f2bf(float x) {
  __bf16 h = (__bf16)x;               // native v_cvt on gfx1250
  return __builtin_bit_cast(u16t, h);
}

__device__ __forceinline__ f32x8 zero8() {
  f32x8 z;
#pragma unroll
  for (int i = 0; i < 8; ++i) z[i] = 0.f;
  return z;
}

__device__ __forceinline__ f32x8 wmma_bf16(bf16x16 a, bf16x16 b, f32x8 c) {
  return __builtin_amdgcn_wmma_f32_16x16x32_bf16(false, a, false, b, (short)0, c,
                                                 false, false);
}

// CDNA5 async direct global->LDS copy (16B per lane), tracked by ASYNCcnt.
// VDST = LDS byte offset (low 32 bits of the flat shared pointer), VADDR = 64b addr.
__device__ __forceinline__ void async_g2l_b128(u16t* ldst, const u16t* gsrc) {
  unsigned loff = (unsigned)(uintptr_t)ldst;
  asm volatile("global_load_async_to_lds_b128 %0, %1, off"
               :: "v"(loff), "v"(gsrc) : "memory");
}
__device__ __forceinline__ void wait_async0() {
  asm volatile("s_wait_asynccnt 0" ::: "memory");
}

// A-fragment from row-major [16 x 32] bf16 tile (global or LDS), row stride rs.
__device__ __forceinline__ bf16x16 load_a_rm(const u16t* base, int rowStride, int lane) {
  int m = lane & 15, g = lane >> 4;
  const u16t* p = base + (size_t)m * rowStride + g * 8;
  Frag f;
  f.q[0] = *(const uint4*)(p);
  f.q[1] = *(const uint4*)(p + 16);
  return f.v;
}

// B-fragment from K-major source: column n = lane % 16, k in table order (8g+s, 16+8g+s).
__device__ __forceinline__ bf16x16 load_b_rm(const u16t* base, int rowStride, int lane) {
  int n = lane & 15, g = lane >> 4;
  Frag f;
#pragma unroll
  for (int s = 0; s < 8; ++s) {
    f.u[s] = base[(size_t)(8 * g + s) * rowStride + n];
    f.u[8 + s] = base[(size_t)(16 + 8 * g + s) * rowStride + n];
  }
  return f.v;
}

__device__ __forceinline__ float rmax16(float v) {
  v = fmaxf(v, __shfl_xor(v, 1, 16));
  v = fmaxf(v, __shfl_xor(v, 2, 16));
  v = fmaxf(v, __shfl_xor(v, 4, 16));
  v = fmaxf(v, __shfl_xor(v, 8, 16));
  return v;
}
__device__ __forceinline__ float rsum16(float v) {
  v += __shfl_xor(v, 1, 16);
  v += __shfl_xor(v, 2, 16);
  v += __shfl_xor(v, 4, 16);
  v += __shfl_xor(v, 8, 16);
  return v;
}

// ---------------- prep kernels ----------------
__global__ __launch_bounds__(256) void k_pack_corr(const float* __restrict__ corr,
                                                   u16t* __restrict__ corrT) {
  __shared__ float tile[64][WPAD + 1];
  int n = blockIdx.x / HPAD;
  int ho = blockIdx.x % HPAD;
  int hs = ho - 1;
  bool hok = (unsigned)hs < (unsigned)HH;
  int tid = threadIdx.x;
  u16t* dst = corrT + ((size_t)(n * HPAD + ho)) * WPAD * CPAD;
  for (int c0 = 0; c0 < CPAD; c0 += 64) {
    for (int e = tid; e < 64 * WPAD; e += 256) {
      int cr = e / WPAD, wo = e % WPAD;
      int c = c0 + cr;
      int wsrc = wo - 1;
      float v = 0.f;
      if (hok && c < CIN && (unsigned)wsrc < (unsigned)WWID)
        v = corr[(((size_t)n * CIN + c) * HH + hs) * WWID + wsrc];
      tile[cr][wo] = v;
    }
    __syncthreads();
    for (int e = tid; e < 64 * WPAD; e += 256) {
      int wo = e / 64, cr = e % 64;
      int c = c0 + cr;
      if (c < CPAD) dst[(size_t)wo * CPAD + c] = f2bf(tile[cr][wo]);
    }
    __syncthreads();
  }
}

__global__ void k_pack_wsk(const float* __restrict__ w, u16t* __restrict__ wB) {
  int i = blockIdx.x * 256 + threadIdx.x;
  if (i >= KTAPS * CPAD * DCH) return;
  int d = i & 127;
  int kk = i >> 7;
  int tap = kk / CPAD;
  int cp = kk % CPAD;
  float v = (cp < CIN) ? w[(size_t)d * (CIN * 9) + (size_t)cp * 9 + tap] : 0.f;
  wB[i] = f2bf(v);
}

__global__ void k_pack_w1x1(const float* __restrict__ w, u16t* __restrict__ o, int K) {
  int i = blockIdx.x * 256 + threadIdx.x;
  if (i >= K * DCH) return;
  int d = i & 127;
  int kk = i >> 7;
  o[i] = f2bf(w[(size_t)d * K + kk]);
}

__global__ void k_cvt_bf16(const float* __restrict__ src, u16t* __restrict__ dst, int total) {
  int i = blockIdx.x * 256 + threadIdx.x;
  if (i < total) dst[i] = f2bf(src[i]);
}

__global__ void k_cvt_v_tr(const float* __restrict__ v, u16t* __restrict__ vf) {
  int i = blockIdx.x * 256 + threadIdx.x;
  if (i >= NB * HWP * DCH) return;
  int d = i & 127;
  int p = (i >> 7) % HWP;
  int n = i / (HWP * DCH);
  vf[i] = f2bf(v[((size_t)n * DCH + d) * HWP + p]);
}

// ---------------- kernel 1: sk_conv (implicit GEMM; 8 waves share W via LDS) ----------------
__global__ __launch_bounds__(256) void k_conv_sk(
    const u16t* __restrict__ corrT, const u16t* __restrict__ wB,
    const float* __restrict__ b_sk, float* __restrict__ resF,
    u16t* __restrict__ xcat, u16t* __restrict__ qB) {
  __shared__ u16t ldsW[2][32 * DROW];   // double-buffered 32x128 weight tile
  int tid = threadIdx.x;
  int lane = tid & 31;
  int wv = tid >> 5;
  int tile = blockIdx.x * 8 + wv;      // 0..1535
  int n = tile / 192;
  int pb = (tile % 192) * 16;
  int h = pb / WWID;
  int w0 = pb % WWID;
  int m = lane & 15, g = lane >> 4;

  const u16t* cbase = corrT + (size_t)n * HPAD * WPAD * CPAD;
  auto a_ptr = [&](int ks) {
    int tap = ks / 11, cbi = ks % 11;
    return cbase + ((size_t)(h + tap / 3) * WPAD + (w0 + tap % 3)) * CPAD + cbi * 32;
  };
  // cooperative async W copy: 32x128 bf16 = 512 16B chunks, 2 per thread
  auto wcopy = [&](int ks, int buf) {
#pragma unroll
    for (int j = 0; j < 2; ++j) {
      int c = tid + 256 * j;
      int row = c >> 4, part = c & 15;
      async_g2l_b128(&ldsW[buf][row * DROW + part * 8],
                     wB + (size_t)ks * 32 * DCH + row * DCH + part * 8);
    }
  };

  f32x8 acc[8];
#pragma unroll
  for (int t = 0; t < 8; ++t) acc[t] = zero8();

  wcopy(0, 0);
  wait_async0();
  __syncthreads();
  bf16x16 a = load_a_rm(a_ptr(0), CPAD, lane);

  for (int ks = 0; ks < 99; ++ks) {
    bool more = (ks + 1 < 99);
    if (more) wcopy(ks + 1, (ks + 1) & 1);   // async copy overlaps the WMMAs
    bf16x16 an = more ? load_a_rm(a_ptr(ks + 1), CPAD, lane) : a;
    const u16t* lw = &ldsW[ks & 1][0];
    bf16x16 bs[8];
#pragma unroll
    for (int t = 0; t < 8; ++t) bs[t] = load_b_rm(lw + t * 16, DROW, lane);
#pragma unroll
    for (int t = 0; t < 8; ++t) acc[t] = wmma_bf16(a, bs[t], acc[t]);
    a = an;
    wait_async0();
    __syncthreads();
  }

  const float scale = 0.08838834764831845f;  // 128^-0.5 folded into Q
#pragma unroll
  for (int t = 0; t < 8; ++t) {
    int d = t * 16 + m;
    float bias = b_sk[d];
#pragma unroll
    for (int r = 0; r < 8; ++r) {
      int p = pb + 8 * g + r;
      float q = acc[t][r] + bias;
      size_t pd = (size_t)n * HWP + p;
      resF[pd * DCH + d] = q;
      xcat[pd * 256 + DCH + d] = f2bf(q);
      qB[pd * DCH + d] = f2bf(q * scale);
    }
  }
}

// ---------------- kernel 2: flash attention (8 waves share K/V tiles via LDS) ----------------
__global__ __launch_bounds__(256) void k_attn(
    const u16t* __restrict__ qB, const u16t* __restrict__ kB,
    const u16t* __restrict__ vfB, u16t* __restrict__ xcat) {
  __shared__ u16t ldsK[2][DCH * KROW];    // [d][key] keys 0..31, padded rows
  __shared__ u16t ldsV[2][32 * DROW];     // [key][d], padded rows
  __shared__ u16t ldsP[8][16 * KROW];     // per-wave P tile
  int tid = threadIdx.x;
  int lane = tid & 31;
  int wv = tid >> 5;
  int blk = blockIdx.x;                   // 0..191
  int n = blk / 24;
  int pb = ((blk % 24) * 8 + wv) * 16;
  int m = lane & 15, g = lane >> 4;

  const u16t* kbase = kB + (size_t)n * DCH * HWP;
  const u16t* vbase = vfB + (size_t)n * HWP * DCH;

  // cooperative async K+V copy: 8KB each = 512 chunks each, 2+2 per thread
  auto kvcopy = [&](int kb, int buf) {
#pragma unroll
    for (int j = 0; j < 2; ++j) {
      int c = tid + 256 * j;
      int d = c >> 2, part = c & 3;       // 4 chunks per 32-key row
      async_g2l_b128(&ldsK[buf][d * KROW + part * 8],
                     kbase + (size_t)d * HWP + kb + part * 8);
    }
#pragma unroll
    for (int j = 0; j < 2; ++j) {
      int c = tid + 256 * j;
      int key = c >> 4, part = c & 15;    // 16 chunks per 128-d row
      async_g2l_b128(&ldsV[buf][key * DROW + part * 8],
                     vbase + (size_t)(kb + key) * DCH + part * 8);
    }
  };

  kvcopy(0, 0);

  bf16x16 Q[4];
  const u16t* qbase = qB + ((size_t)n * HWP + pb) * DCH;
#pragma unroll
  for (int ks = 0; ks < 4; ++ks) Q[ks] = load_a_rm(qbase + ks * 32, DCH, lane);

  float mrow[8], lrow[8];
  f32x8 O[8];
#pragma unroll
  for (int r = 0; r < 8; ++r) { mrow[r] = -1e30f; lrow[r] = 0.f; }
#pragma unroll
  for (int t = 0; t < 8; ++t) O[t] = zero8();

  u16t* lp = &ldsP[wv][0];
  wait_async0();
  __syncthreads();

  for (int i = 0; i < HWP / 32; ++i) {
    bool more = (i + 1 < HWP / 32);
    if (more) kvcopy((i + 1) * 32, (i + 1) & 1);  // async, overlaps compute
    const u16t* lk = &ldsK[i & 1][0];
    const u16t* lv = &ldsV[i & 1][0];

    // ---- S = Q @ K^T ----
    bf16x16 bk[8];
#pragma unroll
    for (int ks = 0; ks < 4; ++ks) {
      bk[2 * ks] = load_b_rm(lk + (size_t)(ks * 32) * KROW, KROW, lane);
      bk[2 * ks + 1] = load_b_rm(lk + (size_t)(ks * 32) * KROW + 16, KROW, lane);
    }
    f32x8 S0 = zero8(), S1 = zero8();
#pragma unroll
    for (int ks = 0; ks < 4; ++ks) {
      S0 = wmma_bf16(Q[ks], bk[2 * ks], S0);
      S1 = wmma_bf16(Q[ks], bk[2 * ks + 1], S1);
    }
    // ---- online softmax ----
    float cfa[8];
#pragma unroll
    for (int r = 0; r < 8; ++r) {
      float mx = rmax16(fmaxf(S0[r], S1[r]));
      float mn = fmaxf(mrow[r], mx);
      float cf = __expf(mrow[r] - mn);
      float p0 = __expf(S0[r] - mn);
      float p1 = __expf(S1[r] - mn);
      lrow[r] = lrow[r] * cf + rsum16(p0 + p1);
      mrow[r] = mn;
      cfa[r] = cf;
      int row = 8 * g + r;
      lp[row * KROW + m] = f2bf(p0);
      lp[row * KROW + 16 + m] = f2bf(p1);
    }
#pragma unroll
    for (int r = 0; r < 8; ++r) {
      float cf = cfa[r];
#pragma unroll
      for (int t = 0; t < 8; ++t) O[t][r] *= cf;
    }
    asm volatile("s_wait_dscnt 0" ::: "memory");   // P stores -> A-frag loads
    bf16x16 P = load_a_rm(lp, KROW, lane);
    // ---- O += P @ V ----
    bf16x16 bv[8];
#pragma unroll
    for (int t = 0; t < 8; ++t) bv[t] = load_b_rm(lv + t * 16, DROW, lane);
#pragma unroll
    for (int t = 0; t < 8; ++t) O[t] = wmma_bf16(P, bv[t], O[t]);

    wait_async0();
    __syncthreads();
  }
#pragma unroll
  for (int t = 0; t < 8; ++t) {
    int d = t * 16 + m;
#pragma unroll
    for (int r = 0; r < 8; ++r) {
      int p = pb + 8 * g + r;
      float o = O[t][r] / lrow[r];
      xcat[((size_t)n * HWP + p) * 256 + d] = f2bf(o);
    }
  }
}

// ---------------- kernel 3: proj + residual + FFN (3 chained GEMMs) ----------------
__global__ __launch_bounds__(128) void k_mlp(
    const u16t* __restrict__ xcat, const float* __restrict__ resF,
    const u16t* __restrict__ wPB, const float* __restrict__ b_proj,
    const u16t* __restrict__ wF1, const float* __restrict__ b_ffn1,
    const u16t* __restrict__ wF2, const float* __restrict__ b_ffn2,
    float* __restrict__ out) {
  __shared__ u16t ldsX[4][16 * DROW];
  __shared__ u16t ldsH[4][16 * DROW];
  int lane = threadIdx.x & 31;
  int wv = threadIdx.x >> 5;
  int tile = blockIdx.x * 4 + wv;
  int n = tile / 192;
  int pb = (tile % 192) * 16;
  int m = lane & 15, g = lane >> 4;

  // GEMM1: x = concat(attn,res) @ wProj^T   (K=256), rotating prefetch
  f32x8 acc[8];
#pragma unroll
  for (int t = 0; t < 8; ++t) acc[t] = zero8();
  const u16t* xbase = xcat + ((size_t)n * HWP + pb) * 256;
  {
    bf16x16 a = load_a_rm(xbase, 256, lane);
    bf16x16 bs[8];
#pragma unroll
    for (int t = 0; t < 8; ++t) bs[t] = load_b_rm(wPB + t * 16, DCH, lane);
    for (int kb = 0; kb < 256; kb += 32) {
      int nx = (kb + 32 < 256) ? kb + 32 : kb;
      bf16x16 an = load_a_rm(xbase + nx, 256, lane);
      const u16t* wrow = wPB + (size_t)nx * DCH;
#pragma unroll
      for (int t = 0; t < 8; ++t) {
        acc[t] = wmma_bf16(a, bs[t], acc[t]);
        bs[t] = load_b_rm(wrow + t * 16, DCH, lane);
      }
      a = an;
    }
  }
  float xv[8][8];
  u16t* lx = &ldsX[wv][0];
#pragma unroll
  for (int t = 0; t < 8; ++t) {
    int d = t * 16 + m;
    float bias = b_proj[d];
#pragma unroll
    for (int r = 0; r < 8; ++r) {
      int p = pb + 8 * g + r;
      float x = acc[t][r] + bias + resF[((size_t)n * HWP + p) * DCH + d];
      xv[t][r] = x;
      lx[(8 * g + r) * DROW + d] = f2bf(x);
    }
  }
  asm volatile("s_wait_dscnt 0" ::: "memory");

  // GEMM2: h = gelu(x @ wFFN1^T + b1)   (K=128)
#pragma unroll
  for (int t = 0; t < 8; ++t) acc[t] = zero8();
  {
    bf16x16 a = load_a_rm(lx, DROW, lane);
    bf16x16 bs[8];
#pragma unroll
    for (int t = 0; t < 8; ++t) bs[t] = load_b_rm(wF1 + t * 16, DCH, lane);
    for (int kb = 0; kb < DCH; kb += 32) {
      int nx = (kb + 32 < DCH) ? kb + 32 : kb;
      bf16x16 an = load_a_rm(lx + nx, DROW, lane);
      const u16t* wrow = wF1 + (size_t)nx * DCH;
#pragma unroll
      for (int t = 0; t < 8; ++t) {
        acc[t] = wmma_bf16(a, bs[t], acc[t]);
        bs[t] = load_b_rm(wrow + t * 16, DCH, lane);
      }
      a = an;
    }
  }
  u16t* lh = &ldsH[wv][0];
#pragma unroll
  for (int t = 0; t < 8; ++t) {
    int d = t * 16 + m;
    float bias = b_ffn1[d];
#pragma unroll
    for (int r = 0; r < 8; ++r) {
      float hc = acc[t][r] + bias;
      hc = 0.5f * hc * (1.f + erff(hc * 0.70710678118654752f));
      lh[(8 * g + r) * DROW + d] = f2bf(hc);
    }
  }
  asm volatile("s_wait_dscnt 0" ::: "memory");

  // GEMM3: out = x + h @ wFFN2^T + b2   (K=128)
#pragma unroll
  for (int t = 0; t < 8; ++t) acc[t] = zero8();
  {
    bf16x16 a = load_a_rm(lh, DROW, lane);
    bf16x16 bs[8];
#pragma unroll
    for (int t = 0; t < 8; ++t) bs[t] = load_b_rm(wF2 + t * 16, DCH, lane);
    for (int kb = 0; kb < DCH; kb += 32) {
      int nx = (kb + 32 < DCH) ? kb + 32 : kb;
      bf16x16 an = load_a_rm(lh + nx, DROW, lane);
      const u16t* wrow = wF2 + (size_t)nx * DCH;
#pragma unroll
      for (int t = 0; t < 8; ++t) {
        acc[t] = wmma_bf16(a, bs[t], acc[t]);
        bs[t] = load_b_rm(wrow + t * 16, DCH, lane);
      }
      a = an;
    }
  }
#pragma unroll
  for (int t = 0; t < 8; ++t) {
    int d = t * 16 + m;
    float bias = b_ffn2[d];
#pragma unroll
    for (int r = 0; r < 8; ++r) {
      int p = pb + 8 * g + r;
      out[((size_t)n * DCH + d) * HWP + p] = xv[t][r] + acc[t][r] + bias;
    }
  }
}

// ---------------- host ----------------
extern "C" void kernel_launch(void* const* d_in, const int* in_sizes, int n_in,
                              void* d_out, int out_size, void* d_ws, size_t ws_size,
                              hipStream_t stream) {
  const float* corr = (const float*)d_in[0];
  const float* kin  = (const float*)d_in[1];
  const float* vin  = (const float*)d_in[2];
  const float* w_sk = (const float*)d_in[3];
  const float* b_sk = (const float*)d_in[4];
  const float* w_pr = (const float*)d_in[5];
  const float* b_pr = (const float*)d_in[6];
  const float* w_f1 = (const float*)d_in[7];
  const float* b_f1 = (const float*)d_in[8];
  const float* w_f2 = (const float*)d_in[9];
  const float* b_f2 = (const float*)d_in[10];
  float* out = (float*)d_out;

  char* ws = (char*)d_ws;
  size_t off = 0;
  auto take = [&](size_t bytes) {
    char* p = ws + off;
    off = (off + bytes + 255) & ~(size_t)255;
    return p;
  };
  u16t* corrT = (u16t*)take((size_t)NB * HPAD * WPAD * CPAD * 2);
  u16t* wB   = (u16t*)take((size_t)KTAPS * CPAD * DCH * 2);
  u16t* wPB  = (u16t*)take((size_t)256 * DCH * 2);
  u16t* wF1  = (u16t*)take((size_t)DCH * DCH * 2);
  u16t* wF2  = (u16t*)take((size_t)DCH * DCH * 2);
  u16t* kB   = (u16t*)take((size_t)NB * DCH * HWP * 2);
  u16t* vfB  = (u16t*)take((size_t)NB * HWP * DCH * 2);
  u16t* qB   = (u16t*)take((size_t)NB * HWP * DCH * 2);
  float* resF = (float*)take((size_t)NB * HWP * DCH * 4);
  u16t* xcat = (u16t*)take((size_t)NB * HWP * 256 * 2);

  k_pack_corr<<<NB * HPAD, 256, 0, stream>>>(corr, corrT);
  int tw = KTAPS * CPAD * DCH;
  k_pack_wsk<<<(tw + 255) / 256, 256, 0, stream>>>(w_sk, wB);
  k_pack_w1x1<<<(256 * DCH + 255) / 256, 256, 0, stream>>>(w_pr, wPB, 256);
  k_pack_w1x1<<<(DCH * DCH + 255) / 256, 256, 0, stream>>>(w_f1, wF1, DCH);
  k_pack_w1x1<<<(DCH * DCH + 255) / 256, 256, 0, stream>>>(w_f2, wF2, DCH);
  int tk = NB * DCH * HWP;
  k_cvt_bf16<<<(tk + 255) / 256, 256, 0, stream>>>(kin, kB, tk);
  k_cvt_v_tr<<<(tk + 255) / 256, 256, 0, stream>>>(vin, vfB);

  k_conv_sk<<<192, 256, 0, stream>>>(corrT, wB, b_sk, resF, xcat, qB);
  k_attn<<<192, 256, 0, stream>>>(qB, kB, vfB, xcat);
  k_mlp<<<384, 128, 0, stream>>>(xcat, resF, wPB, b_pr, wF1, b_f1, wF2, b_f2, out);

  (void)in_sizes; (void)n_in; (void)out_size; (void)ws_size;
}
